// Net_38646115729829
// MI455X (gfx1250) — matrix operand order repaired and checked by reading, not compiled, not run on hardware
//
#include <hip/hip_runtime.h>
#include <hip/hip_bf16.h>

typedef __attribute__((ext_vector_type(16))) __bf16 v16bf;
typedef __attribute__((ext_vector_type(4)))  __bf16 v4bf;
typedef __attribute__((ext_vector_type(8)))  float  v8f;

#define N_NODES 100000
#define N_FEAT  1433
#define HID     64
#define NCLS    7

// ---------------- degree / dinv ----------------
__global__ void k_deg_init(float* __restrict__ deg, int n) {
    int i = blockIdx.x * blockDim.x + threadIdx.x;
    if (i < n) deg[i] = 1.0f;                       // self-loop contributes 1
}

__global__ void k_deg_edges(const int* __restrict__ dst, float* __restrict__ deg, int nE) {
    int e = blockIdx.x * blockDim.x + threadIdx.x;
    if (e < nE) atomicAdd(&deg[dst[e]], 1.0f);
}

__global__ void k_dinv(float* __restrict__ deg, int n) {
    int i = blockIdx.x * blockDim.x + threadIdx.x;
    if (i < n) {
        float d = deg[i];
        deg[i] = (d > 0.0f) ? rsqrtf(d) : 0.0f;
    }
}

// ---------------- weight pre-pack: W[K][64] fp32 -> fragment-ready bf16 ----------------
// P[kc][ntile][lane][e] : lane = half*16 + l16 ; col = ntile*16 + l16 ; k = kc*32 + half*16 + e
// Each WMMA B fragment is then 32 contiguous bytes per lane (2x global_load_b128, L2-resident).
__global__ void k_pack_w(const float* __restrict__ W, __bf16* __restrict__ P, int K, int nkc) {
    int t = blockIdx.x * blockDim.x + threadIdx.x;   // nkc*1024 threads, 2 elems each
    if (t >= nkc * 1024) return;
    int e2    = (t & 7) * 2;
    int lane  = (t >> 3) & 31;
    int ntile = (t >> 8) & 3;
    int kc    = t >> 10;
    int half  = lane >> 4, l16 = lane & 15;
    int col   = ntile * 16 + l16;
    #pragma unroll
    for (int j = 0; j < 2; ++j) {
        int e = e2 + j;
        int k = kc * 32 + half * 16 + e;
        float v = (k < K) ? W[(long)k * 64 + col] : 0.0f;
        P[((((long)kc * 4 + ntile) * 32 + lane) * 16) + e] = (__bf16)v;
    }
}

// ---------------- bf16 WMMA GEMM: C[M x 64] = A[M x K] * B[K x 64] ----------------
// block = 256 threads = 8 waves; block tile 32 rows x 64 cols, one 16x16 WMMA tile per wave.
// A staged double-buffered in LDS in fragment-packed order (2x ds_load_b128 per fragment).
// B read directly from pre-packed global (2x global_load_b128 per fragment, hits L2).
// mode 0: C = acc
// mode 1: C = acc ; O = acc * dinv[row]^2   (GCN self-loop init; O may alias A's rows)
// mode 2: C = relu(acc + bias[col])
__global__ __launch_bounds__(256)
void k_gemm_bf16_n64(const float* __restrict__ A, const __bf16* __restrict__ Bp,
                     float* __restrict__ C, float* __restrict__ O,
                     const float* __restrict__ dinv, const float* __restrict__ bias,
                     int M, int K, int mode) {
    __shared__ __bf16 Ap[2][2][32][16];   // [buf][mtile][lane][elem] = 2KB x 2

    const int tid   = threadIdx.x;
    const int lane  = tid & 31;
    const int wave  = tid >> 5;
    const int mtile = wave >> 2;      // 0..1
    const int ntile = wave & 3;       // 0..3
    const int mBase = blockIdx.x * 32;
    const int half  = lane >> 4;
    const int l16   = lane & 15;
    const int nkc   = (K + 31) >> 5;

    // ---- staging decode: this thread loads 4 consecutive k of one row, stores one packed b64
    const int sr    = tid >> 3;                         // row in tile 0..31
    const int skk   = (tid & 7) * 4;                    // k offset 0,4,...,28
    const int sHalf = skk >> 4;
    const int sSlot = skk & 15;                         // packed elem start (mult of 4)
    const int sLane = sHalf * 16 + (sr & 15);
    const int sMt   = sr >> 4;
    const bool rowOk = (mBase + sr) < M;
    const long aRow = (long)(mBase + sr) * K;

    auto stage = [&](int kc, int buf) {
        int kb = kc << 5;
        float f0 = 0.f, f1 = 0.f, f2 = 0.f, f3 = 0.f;
        if (rowOk) {
            const float* p = A + aRow + kb + skk;
            if (kb + 32 <= K) {                         // fast path: no guards, 4 loads, 1 wait
                f0 = p[0]; f1 = p[1]; f2 = p[2]; f3 = p[3];
            } else {                                    // K tail (once per block)
                int k0 = kb + skk;
                if (k0 + 0 < K) f0 = p[0];
                if (k0 + 1 < K) f1 = p[1];
                if (k0 + 2 < K) f2 = p[2];
                if (k0 + 3 < K) f3 = p[3];
            }
        }
        v4bf v = { (__bf16)f0, (__bf16)f1, (__bf16)f2, (__bf16)f3 };
        *(v4bf*)&Ap[buf][sMt][sLane][sSlot] = v;        // one ds_store_b64
    };

    v8f acc = {};
    stage(0, 0);
    __syncthreads();

    for (int kc = 0; kc < nkc; ++kc) {
        const int buf = kc & 1;
        if (kc + 1 < nkc) stage(kc + 1, buf ^ 1);       // prefetch next chunk into other buffer

        v16bf af = *(const v16bf*)&Ap[buf][mtile][lane][0];                  // 2x ds_load_b128
        v16bf bf = *(const v16bf*)&Bp[(((long)kc * 4 + ntile) * 32 + lane) * 16]; // 2x b128 (L2)

        acc = __builtin_amdgcn_wmma_f32_16x16x32_bf16(
            false, af, false, bf, (short)0, acc, false, false);

        __syncthreads();                                // 1 barrier per k-step
    }

    // ---- epilogue: lanes 0-15 -> M = v, lanes 16-31 -> M = v+8 ; N = lane%16
    const int col = ntile * 16 + l16;
    #pragma unroll
    for (int v = 0; v < 8; ++v) {
        int row = mBase + mtile * 16 + half * 8 + v;
        if (row >= M) continue;
        float a = acc[v];
        if (mode == 2) {
            C[(long)row * 64 + col] = fmaxf(a + bias[col], 0.0f);
        } else {
            C[(long)row * 64 + col] = a;
            if (mode == 1) {
                float di = dinv[row];
                O[(long)row * 64 + col] = a * di * di;  // self-loop term
            }
        }
    }
}

// ---------------- edge aggregation: out[dst] += h[src] * dinv[src]*dinv[dst] ----------------
__global__ void k_aggr_edges(const int* __restrict__ src, const int* __restrict__ dst,
                             const float* __restrict__ h, const float* __restrict__ dinv,
                             float* __restrict__ out, int nE) {
    int t = blockIdx.x * blockDim.x + threadIdx.x;      // nE*16 threads, 16 lanes/edge
    if (t >= nE * 16) return;
    int e = t >> 4, q = t & 15;
    int s = src[e], d = dst[e];
    float norm = dinv[s] * dinv[d];
    float4 v = ((const float4*)(h + (long)s * 64))[q];
    float* op = out + (long)d * 64 + q * 4;
    atomicAdd(op + 0, v.x * norm);
    atomicAdd(op + 1, v.y * norm);
    atomicAdd(op + 2, v.z * norm);
    atomicAdd(op + 3, v.w * norm);
}

__global__ void k_bias_relu(float* __restrict__ y, const float* __restrict__ b, int n) {
    int t = blockIdx.x * blockDim.x + threadIdx.x;      // n*16 threads
    if (t >= n * 16) return;
    int i = t >> 4, q = t & 15;
    float4 v  = ((float4*)(y + (long)i * 64))[q];
    float4 bb = ((const float4*)b)[q];
    v.x = fmaxf(v.x + bb.x, 0.0f);
    v.y = fmaxf(v.y + bb.y, 0.0f);
    v.z = fmaxf(v.z + bb.z, 0.0f);
    v.w = fmaxf(v.w + bb.w, 0.0f);
    ((float4*)(y + (long)i * 64))[q] = v;
}

// ---------------- final 64->7 projection + log_softmax ----------------
__global__ __launch_bounds__(256)
void k_fc3_logsoftmax(const float* __restrict__ y, const float* __restrict__ W,
                      const float* __restrict__ b, float* __restrict__ out, int n) {
    __shared__ float Ws[64 * NCLS];
    __shared__ float bs[NCLS];
    for (int i = threadIdx.x; i < 64 * NCLS; i += blockDim.x) Ws[i] = W[i];
    if (threadIdx.x < NCLS) bs[threadIdx.x] = b[threadIdx.x];
    __syncthreads();

    int r = blockIdx.x * blockDim.x + threadIdx.x;
    if (r >= n) return;

    float z[NCLS];
    #pragma unroll
    for (int c = 0; c < NCLS; ++c) z[c] = bs[c];

    const float4* yr = (const float4*)(y + (long)r * 64);
    #pragma unroll
    for (int q = 0; q < 16; ++q) {
        float4 v = yr[q];
        int j = q * 4;
        #pragma unroll
        for (int c = 0; c < NCLS; ++c) {
            z[c] += v.x * Ws[(j + 0) * NCLS + c];
            z[c] += v.y * Ws[(j + 1) * NCLS + c];
            z[c] += v.z * Ws[(j + 2) * NCLS + c];
            z[c] += v.w * Ws[(j + 3) * NCLS + c];
        }
    }

    float m = z[0];
    #pragma unroll
    for (int c = 1; c < NCLS; ++c) m = fmaxf(m, z[c]);
    float s = 0.0f;
    #pragma unroll
    for (int c = 0; c < NCLS; ++c) s += __expf(z[c] - m);
    float lse = __logf(s);
    float* orow = out + (long)r * NCLS;
    #pragma unroll
    for (int c = 0; c < NCLS; ++c) orow[c] = z[c] - m - lse;
}

// ---------------- launcher ----------------
extern "C" void kernel_launch(void* const* d_in, const int* in_sizes, int n_in,
                              void* d_out, int out_size, void* d_ws, size_t ws_size,
                              hipStream_t stream) {
    const float* x   = (const float*)d_in[0];
    const int*   ei  = (const int*)  d_in[1];
    const float* W1  = (const float*)d_in[2];
    const float* b1  = (const float*)d_in[3];
    const float* W2  = (const float*)d_in[4];
    const float* b2  = (const float*)d_in[5];
    const float* Wf1 = (const float*)d_in[6];
    const float* bf1 = (const float*)d_in[7];
    const float* Wf2 = (const float*)d_in[8];
    const float* bf2 = (const float*)d_in[9];
    const float* Wf3 = (const float*)d_in[10];
    const float* bf3 = (const float*)d_in[11];
    float* out = (float*)d_out;

    const int N = N_NODES;
    const int E = in_sizes[1] / 2;            // edge_index is [2, E] flat
    const int* src = ei;
    const int* dst = ei + E;
    const int nkc1 = (N_FEAT + 31) / 32;      // 45

    // workspace layout (all chunks 256B-aligned)
    char* w = (char*)d_ws;
    float*  bufA = (float*)w;   w += (size_t)N * 64 * 4;          // 25.6 MB
    float*  bufB = (float*)w;   w += (size_t)N * 64 * 4;          // 25.6 MB
    float*  dinv = (float*)w;   w += ((size_t)N * 4 + 255) & ~(size_t)255;
    __bf16* pW1  = (__bf16*)w;  w += (size_t)nkc1 * 2048 * 2;     // 180 KB
    __bf16* pW2  = (__bf16*)w;  w += (size_t)2 * 2048 * 2;
    __bf16* pWf1 = (__bf16*)w;  w += (size_t)2 * 2048 * 2;
    __bf16* pWf2 = (__bf16*)w;  w += (size_t)2 * 2048 * 2;

    const int T = 256;
    const int gN   = (N + T - 1) / T;
    const int gE   = (E + T - 1) / T;
    const int gN16 = (N * 16 + T - 1) / T;
    const int gE16 = (E * 16 + T - 1) / T;
    const int gM   = (N + 31) / 32;

    // pre-pack all WMMA weights (fragment-ready bf16; stays hot in 192MB L2)
    k_pack_w<<<(nkc1 * 1024 + T - 1) / T, T, 0, stream>>>(W1,  pW1,  N_FEAT, nkc1);
    k_pack_w<<<(2 * 1024 + T - 1) / T,    T, 0, stream>>>(W2,  pW2,  HID, 2);
    k_pack_w<<<(2 * 1024 + T - 1) / T,    T, 0, stream>>>(Wf1, pWf1, HID, 2);
    k_pack_w<<<(2 * 1024 + T - 1) / T,    T, 0, stream>>>(Wf2, pWf2, HID, 2);

    // symmetric degree normalization
    k_deg_init <<<gN, T, 0, stream>>>(dinv, N);
    k_deg_edges<<<gE, T, 0, stream>>>(dst, dinv, E);
    k_dinv     <<<gN, T, 0, stream>>>(dinv, N);

    // GCN layer 1: H1 = X @ W1 (bufA) + self-loop init into bufB; edges; relu(+b1)
    k_gemm_bf16_n64<<<gM, T, 0, stream>>>(x, pW1, bufA, bufB, dinv, nullptr, N, N_FEAT, 1);
    k_aggr_edges   <<<gE16, T, 0, stream>>>(src, dst, bufA, dinv, bufB, E);
    k_bias_relu    <<<gN16, T, 0, stream>>>(bufB, b1, N);

    // GCN layer 2 (O aliases input rows: safe, each block rewrites only rows it alone reads)
    k_gemm_bf16_n64<<<gM, T, 0, stream>>>(bufB, pW2, bufA, bufB, dinv, nullptr, N, HID, 1);
    k_aggr_edges   <<<gE16, T, 0, stream>>>(src, dst, bufA, dinv, bufB, E);
    k_bias_relu    <<<gN16, T, 0, stream>>>(bufB, b2, N);

    // FC head with fused bias+relu epilogues
    k_gemm_bf16_n64<<<gM, T, 0, stream>>>(bufB, pWf1, bufA, nullptr, nullptr, bf1, N, HID, 2);
    k_gemm_bf16_n64<<<gM, T, 0, stream>>>(bufA, pWf2, bufB, nullptr, nullptr, bf2, N, HID, 2);

    k_fc3_logsoftmax<<<gN, T, 0, stream>>>(bufB, Wf3, bf3, out, N);
}
